// CompressedSEAModel_89515708383843
// MI455X (gfx1250) — matrix operand order repaired and checked by reading, not compiled, not run on hardware
//
#include <hip/hip_runtime.h>
#include <math.h>

// ---------------------------------------------------------------------------
// Types for CDNA5 WMMA
// ---------------------------------------------------------------------------
typedef __attribute__((ext_vector_type(16))) __bf16 v16bf;
typedef __attribute__((ext_vector_type(8)))  float  v8f;

union FragBF { v16bf v; ushort4 q[4]; };

#define BN_SCALE 0.9999950000374997f   // 1/sqrt(1+1e-5)

__device__ __forceinline__ float seluf(float x) {
  const float sc = 1.0507009873554805f, al = 1.6732632423543772f;
  return x > 0.f ? sc * x : sc * al * (__expf(x) - 1.f);
}
__device__ __forceinline__ float applyact(float x, int act) {
  if (act == 1) return seluf(x * BN_SCALE);        // selu(x*BN)
  if (act == 2) return seluf(x) * BN_SCALE;        // selu(x)*BN
  if (act == 3) return 1.f / (1.f + __expf(-x));   // sigmoid
  return x;
}
__device__ __forceinline__ unsigned short f2bf(float x) {
  unsigned u = __float_as_uint(x);
  unsigned r = u + 0x7fffu + ((u >> 16) & 1u);     // round-to-nearest-even
  return (unsigned short)(r >> 16);
}

// ---------------------------------------------------------------------------
// 1) xm[b,l,f] = mean_t hs[b,l,t,f]
// ---------------------------------------------------------------------------
__global__ void xm_k(const float* __restrict__ hs, float* __restrict__ xm) {
  int bl = blockIdx.x;                         // 0..191 (b*12+l)
  int f  = blockIdx.y * 256 + threadIdx.x;     // 0..767
  const float* p = hs + (size_t)bl * 600 * 768 + f;
  float s = 0.f;
  for (int t = 0; t < 600; ++t) s += p[(size_t)t * 768];
  xm[(size_t)bl * 768 + f] = s * (1.f / 600.f);
}

// ---------------------------------------------------------------------------
// 2) SE gates y[b,l] and attention gates a[b,l]  -> ya[b][0:12]=y, [12:24]=a
// ---------------------------------------------------------------------------
__global__ void gates_k(const float* __restrict__ xm,
                        const float* __restrict__ sew1, const float* __restrict__ sew2,
                        const float* __restrict__ attW, const float* __restrict__ attW1,
                        const float* __restrict__ attW2, float* __restrict__ ya) {
  int b = blockIdx.x, tid = threadIdx.x;
  __shared__ float mu[12], s0[12];
  if (tid < 12) {
    const float* xp = xm + ((size_t)b * 12 + tid) * 768;
    float sm = 0.f, sd = 0.f;
    for (int f = 0; f < 768; ++f) { sm += xp[f]; sd += xp[f] * attW[f]; }
    mu[tid] = sm * (1.f / 768.f);
    s0[tid] = sd / (1.f + __expf(-sd));          // silu
  }
  __syncthreads();
  if (tid == 0) {
    float h1[4], h2[6];
    for (int j = 0; j < 4; ++j) { float s = 0; for (int l = 0; l < 12; ++l) s += mu[l] * sew1[l * 4 + j]; h1[j] = s > 0 ? s : 0; }
    for (int l = 0; l < 12; ++l) { float s = 0; for (int j = 0; j < 4; ++j) s += h1[j] * sew2[j * 12 + l]; ya[b * 24 + l] = 1.f / (1.f + __expf(-s)); }
    for (int j = 0; j < 6; ++j) { float s = 0; for (int l = 0; l < 12; ++l) s += s0[l] * attW1[l * 6 + j]; h2[j] = s / (1.f + __expf(-s)); }
    for (int l = 0; l < 12; ++l) { float s = 0; for (int j = 0; j < 6; ++j) s += h2[j] * attW2[j * 12 + l]; ya[b * 24 + 12 + l] = 1.f / (1.f + __expf(-s)); }
  }
}

// ---------------------------------------------------------------------------
// 3) W2 = proj_w @ ll_w  [9216,64];  bias2[c] = 0.7*(proj_b@ll_w)[c]+ll_b[c]
// ---------------------------------------------------------------------------
__global__ void w2prep_k(const float* __restrict__ projw, const float* __restrict__ llw,
                         float* __restrict__ W2) {
  long idx = (long)blockIdx.x * 256 + threadIdx.x;
  if (idx >= (long)9216 * 64) return;
  int c = idx & 63, k = (int)(idx >> 6);
  const float* pw = projw + (size_t)k * 768;
  float s = 0.f;
  for (int o = 0; o < 768; ++o) s += pw[o] * llw[o * 64 + c];
  W2[idx] = s;
}
__global__ void bias2_k(const float* projb, const float* llw, const float* llb, float* bias2) {
  int c = threadIdx.x; if (c >= 64) return;
  float s = 0.f;
  for (int o = 0; o < 768; ++o) s += projb[o] * llw[o * 64 + c];
  bias2[c] = 0.7f * s + llb[c];
}

// ---------------------------------------------------------------------------
// 4) Per-batch effective bf16 B operand  Bb[b][l*768+f][c]
// ---------------------------------------------------------------------------
__global__ void bbuild_k(const float* __restrict__ W2, const float* __restrict__ llw,
                         const float* __restrict__ ya, unsigned short* __restrict__ Bb) {
  size_t idx = (size_t)blockIdx.x * 256 + threadIdx.x;
  if (idx >= (size_t)16 * 9216 * 64) return;
  int c = (int)(idx & 63);
  int k = (int)((idx >> 6) % 9216);
  int b = (int)(idx / ((size_t)9216 * 64));
  int l = k / 768, f = k % 768;
  float v = 0.7f * ya[b * 24 + 12 + l] * W2[(size_t)(f * 12 + l) * 64 + c]
          + 0.3f * ya[b * 24 + l]      * llw[f * 64 + c];
  Bb[idx] = f2bf(v);
}

// ---------------------------------------------------------------------------
// 5) Fused front-end GEMM with bf16 WMMA:  xlin[b,600,64]
//    per batch: [600 x 9216] (hs, fp32->bf16) @ [9216 x 64] (Bb, bf16)
//    double-buffered LDS pipeline, 2 WMMA per barrier pair
// ---------------------------------------------------------------------------
__global__ void __launch_bounds__(256) gemm_front_k(
    const float* __restrict__ hs, const unsigned short* __restrict__ Bb,
    const float* __restrict__ bias2, float* __restrict__ xlin) {
  __shared__ unsigned short As[2][32 * 32];   // [m 0..31][k 0..31]
  __shared__ unsigned short Bs[2][64 * 32];   // [c 0..63][k 0..31]  (k contiguous)
  const int b = blockIdx.y;
  const int t0 = blockIdx.x * 32;
  const int tid = threadIdx.x;
  const int lane = tid & 31;
  const int wv = tid >> 5;                 // 8 waves
  const int Mt = wv >> 2, Nt = wv & 3;
  const int arow = tid >> 3;               // 0..31
  const int akq = (tid & 7) * 4;
  const int bcq = (tid & 7) * 8;
  int t_a = t0 + arow; if (t_a > 599) t_a = 599;
  const int ks  = (lane >> 4) * 8;         // A lane half (ISA 16-bit A layout)
  const int bks = (lane >> 4) * 8;         // B lane half
  const int ar = Mt * 16 + (lane & 15);
  const int bc = Nt * 16 + (lane & 15);
  const unsigned short* Bbb = Bb + (size_t)b * (9216 * 64);
  v8f acc = {0.f, 0.f, 0.f, 0.f, 0.f, 0.f, 0.f, 0.f};

  auto stage = [&](int ki, int buf) {
    const int l = ki / 24;
    const int f0 = (ki % 24) * 32;
    // A: fp32 -> bf16
    const float* ap = hs + ((((size_t)b * 12 + l) * 600 + t_a) * 768 + (f0 + akq));
    float4 av = *(const float4*)ap;
    __builtin_prefetch(ap + 64, 0, 0);
    ushort4 ah;
    ah.x = f2bf(av.x); ah.y = f2bf(av.y); ah.z = f2bf(av.z); ah.w = f2bf(av.w);
    *(ushort4*)&As[buf][arow * 32 + akq] = ah;
    // B: scatter to k-contiguous-per-column layout
    const unsigned short* bp = Bbb + ((size_t)(ki * 32 + arow)) * 64 + bcq;
    __builtin_prefetch(bp + 2 * 32 * 64, 0, 0);
    ushort4 b0 = ((const ushort4*)bp)[0];
    ushort4 b1 = ((const ushort4*)bp)[1];
    unsigned short* bs = &Bs[buf][0];
    bs[(bcq + 0) * 32 + arow] = b0.x; bs[(bcq + 1) * 32 + arow] = b0.y;
    bs[(bcq + 2) * 32 + arow] = b0.z; bs[(bcq + 3) * 32 + arow] = b0.w;
    bs[(bcq + 4) * 32 + arow] = b1.x; bs[(bcq + 5) * 32 + arow] = b1.y;
    bs[(bcq + 6) * 32 + arow] = b1.z; bs[(bcq + 7) * 32 + arow] = b1.w;
  };
  auto compute = [&](int buf) {
    FragBF fa, fb;
    const unsigned short* as = &As[buf][0];
    const unsigned short* bs = &Bs[buf][0];
    fa.q[0] = *(const ushort4*)&as[ar * 32 + ks + 0];
    fa.q[1] = *(const ushort4*)&as[ar * 32 + ks + 4];
    fa.q[2] = *(const ushort4*)&as[ar * 32 + 16 + ks + 0];
    fa.q[3] = *(const ushort4*)&as[ar * 32 + 16 + ks + 4];
    fb.q[0] = *(const ushort4*)&bs[bc * 32 + bks + 0];
    fb.q[1] = *(const ushort4*)&bs[bc * 32 + bks + 4];
    fb.q[2] = *(const ushort4*)&bs[bc * 32 + 16 + bks + 0];
    fb.q[3] = *(const ushort4*)&bs[bc * 32 + 16 + bks + 4];
    acc = __builtin_amdgcn_wmma_f32_16x16x32_bf16(false, fa.v, false, fb.v,
                                                  (short)0, acc, false, false);
  };

  stage(0, 0);
  __syncthreads();
  #pragma unroll 1
  for (int ki = 0; ki < 288; ki += 2) {
    stage(ki + 1, 1);          // fill buf1 while buf0 is consumed
    compute(0);
    __syncthreads();
    if (ki + 2 < 288) stage(ki + 2, 0);
    compute(1);
    __syncthreads();
  }

  const float bn = bias2[bc];
  const int mb = (lane >> 4) * 8;          // C layout: lanes>=16 hold M=8+r
  float* op = xlin + ((size_t)b * 600 + (t0 + Mt * 16 + mb)) * 64 + bc;
  if (t0 + 32 <= 600) {                    // fast path: whole tile in range
    for (int r = 0; r < 8; ++r) op[(size_t)r * 64] = acc[r] + bn;
  } else {
    for (int r = 0; r < 8; ++r) {
      int t = t0 + Mt * 16 + mb + r;
      if (t < 600) op[(size_t)r * 64] = acc[r] + bn;
    }
  }
}

// ---------------------------------------------------------------------------
// 6) 3x3 stride-3 maxpool over (c,t) + selu  -> p1[B,21,200]
// ---------------------------------------------------------------------------
__global__ void poolselu_k(const float* __restrict__ xlin, float* __restrict__ p1) {
  int idx = blockIdx.x * 256 + threadIdx.x;
  if (idx >= 16 * 21 * 200) return;
  int w = idx % 200; int t = idx / 200; int h = t % 21; int b = t / 21;
  float m = -1e30f;
  for (int j = 0; j < 3; ++j)
    for (int i = 0; i < 3; ++i) {
      float v = xlin[((size_t)b * 600 + (w * 3 + j)) * 64 + (h * 3 + i)];
      if (v > m) m = v;
    }
  p1[idx] = applyact(m, 1);
}

// ---------------------------------------------------------------------------
// Generic conv (kh x 3, pad_w=1), optional residual add, optional activation
// ---------------------------------------------------------------------------
__global__ void conv2d_k(const float* __restrict__ in, const float* __restrict__ wgt,
                         const float* __restrict__ bias, const float* __restrict__ addb,
                         float* __restrict__ out, int B, int cin, int cout,
                         int H, int Hout, int W, int kh, int padtop, int act) {
  long idx = (long)blockIdx.x * 256 + threadIdx.x;
  long total = (long)B * cout * Hout * W;
  if (idx >= total) return;
  int wo = (int)(idx % W); long t = idx / W;
  int ho = (int)(t % Hout); t /= Hout;
  int co = (int)(t % cout); int b = (int)(t / cout);
  float s = bias[co];
  for (int ci = 0; ci < cin; ++ci) {
    const float* ip = in + ((size_t)b * cin + ci) * H * W;
    const float* wp = wgt + ((size_t)(co * cin + ci)) * kh * 3;
    for (int dh = 0; dh < kh; ++dh) {
      int hi = ho - padtop + dh;
      if (hi < 0 || hi >= H) continue;
      for (int dw = 0; dw < 3; ++dw) {
        int wi = wo - 1 + dw;
        if (wi < 0 || wi >= W) continue;
        s += ip[hi * W + wi] * wp[dh * 3 + dw];
      }
    }
  }
  if (addb) s += addb[idx];
  out[idx] = applyact(s, act);
}

__global__ void ewact_k(float* x, long n, int act) {
  long i = (long)blockIdx.x * 256 + threadIdx.x;
  if (i < n) x[i] = applyact(x[i], act);
}

// 1x1 conv over channels
__global__ void chanlin_k(const float* __restrict__ in, const float* __restrict__ wgt,
                          const float* __restrict__ bias, float* __restrict__ out,
                          int B, int cin, int cout, int HW, int act) {
  long idx = (long)blockIdx.x * 256 + threadIdx.x;
  if (idx >= (long)B * cout * HW) return;
  int hw = (int)(idx % HW); long t = idx / HW;
  int o = (int)(t % cout); int b = (int)(t / cout);
  float s = bias[o];
  for (int c = 0; c < cin; ++c) s += in[((size_t)b * cin + c) * HW + hw] * wgt[o * cin + c];
  out[((size_t)b * cout + o) * HW + hw] = applyact(s, act);
}

// softmax over time (axis -1) weighted sum -> e_S[B,21,32] (+pos)
__global__ void es_k(const float* __restrict__ xr, const float* __restrict__ wv,
                     const float* __restrict__ pos, float* __restrict__ eS) {
  int idx = blockIdx.x * 256 + threadIdx.x;
  if (idx >= 16 * 32 * 21) return;
  int h = idx % 21; int t = idx / 21; int c = t % 32; int b = t / 32;
  const float* xb = xr + ((size_t)(b * 32 + c) * 21 + h) * 200;
  const float* wb = wv + ((size_t)(b * 32 + c) * 21 + h) * 200;
  float mx = -1e30f;
  for (int w = 0; w < 200; ++w) if (wb[w] > mx) mx = wb[w];
  float s = 0.f, acc = 0.f;
  for (int w = 0; w < 200; ++w) { float e = __expf(wb[w] - mx); s += e; acc += e * xb[w]; }
  eS[((size_t)b * 21 + h) * 32 + c] = acc / s + pos[h * 32 + c];
}
// softmax over freq (axis -2) weighted sum -> e_T[B,200,32]
__global__ void et_k(const float* __restrict__ xr, const float* __restrict__ wv,
                     float* __restrict__ eT) {
  long idx = (long)blockIdx.x * 256 + threadIdx.x;
  if (idx >= (long)16 * 32 * 200) return;
  int w = (int)(idx % 200); long t = idx / 200; int c = (int)(t % 32); int b = (int)(t / 32);
  const float* xb = xr + (size_t)(b * 32 + c) * 21 * 200 + w;
  const float* wb = wv + (size_t)(b * 32 + c) * 21 * 200 + w;
  float mx = -1e30f;
  for (int h = 0; h < 21; ++h) { float v = wb[(size_t)h * 200]; if (v > mx) mx = v; }
  float s = 0.f, acc = 0.f;
  for (int h = 0; h < 21; ++h) {
    float e = __expf(wb[(size_t)h * 200] - mx); s += e; acc += e * xb[(size_t)h * 200];
  }
  eT[((size_t)b * 200 + w) * 32 + c] = acc / s;
}

// ---------------------------------------------------------------------------
// GAT / HTRG attention kernels
// ---------------------------------------------------------------------------
__global__ void gatatt_k(const float* __restrict__ X, const float* __restrict__ P,
                         const float* __restrict__ Pb, const float* __restrict__ AW,
                         float* __restrict__ att, int B, int N, int D, int Do, float invt) {
  long idx = (long)blockIdx.x * 256 + threadIdx.x;
  if (idx >= (long)B * N * N) return;
  int m = (int)(idx % N); long t = idx / N; int n = (int)(t % N); int b = (int)(t / N);
  const float* xn = X + ((size_t)b * N + n) * D;
  const float* xmp = X + ((size_t)b * N + m) * D;
  float pm[32];
  for (int d = 0; d < D; ++d) pm[d] = xn[d] * xmp[d];
  float acc = 0.f;
  for (int o = 0; o < Do; ++o) {
    float s = Pb[o];
    for (int d = 0; d < D; ++d) s += pm[d] * P[d * Do + o];
    acc += tanhf(s) * AW[o];
  }
  att[idx] = acc * invt;
}
__global__ void htrgatt_k(const float* __restrict__ X, const float* __restrict__ P,
                          const float* __restrict__ Pb, const float* __restrict__ aw11,
                          const float* __restrict__ aw22, const float* __restrict__ aw12,
                          float* __restrict__ att, int B, int N, int D, int Do, int n1,
                          float invt) {
  long idx = (long)blockIdx.x * 256 + threadIdx.x;
  if (idx >= (long)B * N * N) return;
  int m = (int)(idx % N); long t = idx / N; int n = (int)(t % N); int b = (int)(t / N);
  const float* aw = (n < n1) ? ((m < n1) ? aw11 : aw12) : ((m < n1) ? aw12 : aw22);
  const float* xn = X + ((size_t)b * N + n) * D;
  const float* xmp = X + ((size_t)b * N + m) * D;
  float pm[32];
  for (int d = 0; d < D; ++d) pm[d] = xn[d] * xmp[d];
  float acc = 0.f;
  for (int o = 0; o < Do; ++o) {
    float s = Pb[o];
    for (int d = 0; d < D; ++d) s += pm[d] * P[d * Do + o];
    acc += tanhf(s) * aw[o];
  }
  att[idx] = acc * invt;
}
// column softmax (over senders n, axis=-2), in place
__global__ void smcol_k(float* att, int B, int N) {
  int idx = blockIdx.x * 256 + threadIdx.x;
  if (idx >= B * N) return;
  int m = idx % N, b = idx / N;
  float* base = att + (size_t)b * N * N + m;
  float mx = -1e30f;
  for (int n = 0; n < N; ++n) { float v = base[(size_t)n * N]; if (v > mx) mx = v; }
  float s = 0.f;
  for (int n = 0; n < N; ++n) { float v = __expf(base[(size_t)n * N] - mx); base[(size_t)n * N] = v; s += v; }
  float inv = 1.f / s;
  for (int n = 0; n < N; ++n) base[(size_t)n * N] *= inv;
}
__global__ void agg_k(const float* __restrict__ att, const float* __restrict__ X,
                      float* __restrict__ out, int B, int N, int D) {
  long idx = (long)blockIdx.x * 256 + threadIdx.x;
  if (idx >= (long)B * N * D) return;
  int d = (int)(idx % D); long t = idx / D; int n = (int)(t % N); int b = (int)(t / N);
  const float* arow = att + ((size_t)b * N + n) * N;
  float s = 0.f;
  for (int m = 0; m < N; ++m) s += arow[m] * X[((size_t)b * N + m) * D + d];
  out[idx] = s;
}
// out[b,n,o] = act( A@wA + bA + X2@wB + bB ) ; X2 batch stride st2b (broadcast if 0)
__global__ void duallin_k(const float* __restrict__ A, const float* __restrict__ X2,
                          const float* __restrict__ wA, const float* __restrict__ bA,
                          const float* __restrict__ wB, const float* __restrict__ bB,
                          float* __restrict__ out, int B, int N, int D, int O,
                          int st2b, int act) {
  long idx = (long)blockIdx.x * 256 + threadIdx.x;
  if (idx >= (long)B * N * O) return;
  int o = (int)(idx % O); long t = idx / O; int n = (int)(t % N); int b = (int)(t / N);
  const float* a = A + ((size_t)b * N + n) * D;
  const float* x2 = X2 + (size_t)b * st2b + (size_t)n * D;
  float s = bA[o] + bB[o];
  for (int d = 0; d < D; ++d) s += a[d] * wA[d * O + o] + x2[d] * wB[d * O + o];
  out[idx] = applyact(s, act);
}
__global__ void lin_k(const float* __restrict__ X, const float* __restrict__ W,
                      const float* __restrict__ bias, float* __restrict__ out,
                      int B, int N, int D, int O, int Nout, int rowoff, int act) {
  long idx = (long)blockIdx.x * 256 + threadIdx.x;
  if (idx >= (long)B * N * O) return;
  int o = (int)(idx % O); long t = idx / O; int n = (int)(t % N); int b = (int)(t / N);
  const float* x = X + ((size_t)b * N + n) * D;
  float s = bias[o];
  for (int d = 0; d < D; ++d) s += x[d] * W[d * O + o];
  out[((size_t)b * Nout + rowoff + n) * O + o] = applyact(s, act);
}
__global__ void copyrows_k(const float* __restrict__ src, float* __restrict__ dst,
                           int B, int Nsrc, int rowoff, int n, int D) {
  long idx = (long)blockIdx.x * 256 + threadIdx.x;
  if (idx >= (long)B * n * D) return;
  int d = (int)(idx % D); long t = idx / D; int r = (int)(t % n); int b = (int)(t / n);
  dst[idx] = src[((size_t)b * Nsrc + rowoff + r) * D + d];
}
// sigmoid score + top-k (descending, first-index ties) + gather of h*score
__global__ void pool_k(const float* __restrict__ Xh, const float* __restrict__ pw,
                       const float* __restrict__ pb, float* __restrict__ out,
                       int N, int D, int k) {
  int b = blockIdx.x, tid = threadIdx.x;
  __shared__ float sc[256];
  __shared__ float sv[128];
  __shared__ int si[128];
  for (int n = tid; n < N; n += blockDim.x) {
    const float* xp = Xh + ((size_t)b * N + n) * D;
    float s = pb[0];
    for (int d = 0; d < D; ++d) s += xp[d] * pw[d];
    sc[n] = 1.f / (1.f + __expf(-s));
  }
  __syncthreads();
  if (tid == 0) {
    for (int i = 0; i < k; ++i) {
      int bi = 0; float bv = -1e30f;
      for (int n = 0; n < N; ++n) if (sc[n] > bv) { bv = sc[n]; bi = n; }
      si[i] = bi; sv[i] = bv; sc[bi] = -1e30f;
    }
  }
  __syncthreads();
  for (int e = tid; e < k * D; e += blockDim.x) {
    int i = e / D, d = e % D;
    out[((size_t)b * k + i) * D + d] = Xh[((size_t)b * N + si[i]) * D + d] * sv[i];
  }
}
// master attention logits
__global__ void mastatt_k(const float* __restrict__ XC, const float* __restrict__ M,
                          int mbs, const float* __restrict__ PM, const float* __restrict__ PMb,
                          const float* __restrict__ awM, float* __restrict__ am,
                          int B, int N, int D, int Do, float invt) {
  int idx = blockIdx.x * 256 + threadIdx.x;
  if (idx >= B * N) return;
  int n = idx % N, b = idx / N;
  const float* x = XC + ((size_t)b * N + n) * D;
  const float* mm = M + (size_t)b * mbs;
  float pm[32];
  for (int d = 0; d < D; ++d) pm[d] = x[d] * mm[d];
  float acc = 0.f;
  for (int o = 0; o < Do; ++o) {
    float s = PMb[o];
    for (int d = 0; d < D; ++d) s += pm[d] * PM[d * Do + o];
    acc += tanhf(s) * awM[o];
  }
  am[idx] = acc * invt;
}
// softmax over n + weighted sum -> m1[b,D]
__global__ void mastagg_k(const float* __restrict__ am, const float* __restrict__ XC,
                          float* __restrict__ m1, int N, int D) {
  int b = blockIdx.x, tid = threadIdx.x;
  __shared__ float e[128];
  __shared__ float ssum;
  if (tid == 0) {
    float mx = -1e30f;
    for (int n = 0; n < N; ++n) if (am[b * N + n] > mx) mx = am[b * N + n];
    float s = 0.f;
    for (int n = 0; n < N; ++n) { float v = __expf(am[b * N + n] - mx); e[n] = v; s += v; }
    ssum = s;
  }
  __syncthreads();
  float inv = 1.f / ssum;
  for (int d = tid; d < D; d += blockDim.x) {
    float s = 0.f;
    for (int n = 0; n < N; ++n) s += e[n] * XC[((size_t)b * N + n) * D + d];
    m1[b * D + d] = s * inv;
  }
}
__global__ void add_k(const float* a, const float* b, float* o, long n) {
  long i = (long)blockIdx.x * 256 + threadIdx.x; if (i < n) o[i] = a[i] + b[i];
}
__global__ void max_k(const float* a, const float* b, float* o, long n) {
  long i = (long)blockIdx.x * 256 + threadIdx.x; if (i < n) o[i] = fmaxf(a[i], b[i]);
}
// final readout
__global__ void head_k(const float* __restrict__ oT, const float* __restrict__ oS,
                       const float* __restrict__ mst, const float* __restrict__ outw,
                       const float* __restrict__ outb, float* __restrict__ out,
                       int nT, int nS) {
  int b = blockIdx.x;
  if (threadIdx.x != 0) return;
  float last[80];
  for (int o = 0; o < 16; ++o) {
    float ma = -1e30f, sm = 0.f;
    for (int n = 0; n < nT; ++n) { float v = oT[((size_t)b * nT + n) * 16 + o]; float a = fabsf(v); if (a > ma) ma = a; sm += v; }
    last[o] = ma; last[16 + o] = sm / nT;
    ma = -1e30f; sm = 0.f;
    for (int n = 0; n < nS; ++n) { float v = oS[((size_t)b * nS + n) * 16 + o]; float a = fabsf(v); if (a > ma) ma = a; sm += v; }
    last[32 + o] = ma; last[48 + o] = sm / nS;
    last[64 + o] = mst[b * 16 + o];
  }
  float s = outb[0];
  for (int i = 0; i < 80; ++i) s += last[i] * outw[i];
  out[b] = s;
}

// ---------------------------------------------------------------------------
// Host orchestration
// ---------------------------------------------------------------------------
extern "C" void kernel_launch(void* const* d_in, const int* in_sizes, int n_in,
                              void* d_out, int out_size, void* d_ws, size_t ws_size,
                              hipStream_t stream) {
  (void)in_sizes; (void)n_in; (void)out_size; (void)ws_size;
  const int B = 16;
  auto pf = [&](int i) -> const float* { return (const float*)d_in[i]; };
  char* wsp = (char*)d_ws; size_t wo = 0;
  auto alloc = [&](size_t bytes) -> void* { void* p = wsp + wo; wo = (wo + bytes + 255) & ~(size_t)255; return p; };
  auto allocf = [&](size_t n) -> float* { return (float*)alloc(n * 4); };
  auto g = [](long n) { return (unsigned)((n + 255) / 256); };

  float* xm    = allocf((size_t)B * 12 * 768);
  float* ya    = allocf((size_t)B * 24);
  float* W2    = allocf((size_t)9216 * 64);
  float* bias2 = allocf(64);
  unsigned short* Bb = (unsigned short*)alloc((size_t)B * 9216 * 64 * 2);
  float* xlin  = allocf((size_t)B * 600 * 64);
  float* p1    = allocf((size_t)B * 21 * 200);
  float* rb0   = allocf((size_t)B * 32 * 22 * 200);
  float* rb1   = allocf((size_t)B * 32 * 22 * 200);
  float* rb2   = allocf((size_t)B * 32 * 22 * 200);
  float* w64   = allocf((size_t)B * 64 * 21 * 200);
  float* w32b  = allocf((size_t)B * 32 * 21 * 200);
  float* eS    = allocf((size_t)B * 21 * 32);
  float* eT    = allocf((size_t)B * 200 * 32);
  float* attb  = allocf((size_t)B * 200 * 200);
  float* aggb  = allocf((size_t)B * 200 * 32);
  float* ybuf  = allocf((size_t)B * 200 * 32);
  float* gTy   = allocf((size_t)B * 200 * 32);
  float* gSy   = allocf((size_t)B * 21 * 32);
  float* outS  = allocf((size_t)B * 10 * 32);
  float* outT  = allocf((size_t)B * 100 * 32);
  float* xc    = allocf((size_t)B * 110 * 32);
  float* amb   = allocf((size_t)B * 110);
  float* m1b   = allocf((size_t)B * 32);
  float* tA    = allocf((size_t)B * 100 * 16);
  float* sA    = allocf((size_t)B * 10 * 16);
  float* mA    = allocf((size_t)B * 16);
  float* t1p   = allocf((size_t)B * 50 * 16);
  float* s1p   = allocf((size_t)B * 5 * 16);
  float* taB   = allocf((size_t)B * 50 * 16);
  float* saB   = allocf((size_t)B * 5 * 16);
  float* maB   = allocf((size_t)B * 16);
  float* tA2   = allocf((size_t)B * 100 * 16);
  float* sA2   = allocf((size_t)B * 10 * 16);
  float* mA2   = allocf((size_t)B * 16);
  float* t2p   = allocf((size_t)B * 50 * 16);
  float* s2p   = allocf((size_t)B * 5 * 16);
  float* ta2   = allocf((size_t)B * 50 * 16);
  float* sa2   = allocf((size_t)B * 5 * 16);
  float* ma2   = allocf((size_t)B * 16);

  // ---- front-end -----------------------------------------------------------
  xm_k<<<dim3(192, 3), 256, 0, stream>>>(pf(0), xm);
  gates_k<<<B, 64, 0, stream>>>(xm, pf(1), pf(2), pf(3), pf(4), pf(5), ya);
  w2prep_k<<<g((long)9216 * 64), 256, 0, stream>>>(pf(6), pf(8), W2);
  bias2_k<<<1, 64, 0, stream>>>(pf(7), pf(8), pf(9), bias2);
  bbuild_k<<<g((long)B * 9216 * 64), 256, 0, stream>>>(W2, pf(8), ya, Bb);
  gemm_front_k<<<dim3(19, B), 256, 0, stream>>>(pf(0), Bb, bias2, xlin);
  poolselu_k<<<g((long)B * 21 * 200), 256, 0, stream>>>(xlin, p1);

  // ---- residual conv stack -------------------------------------------------
  int bw1[6] = {10, 16, 20, 26, 30, 34}, bb1[6] = {11, 17, 21, 27, 31, 35};
  int bw2[6] = {12, 18, 22, 28, 32, 36}, bb2[6] = {13, 19, 23, 29, 33, 37};
  int bwd[6] = {14, -1, 24, -1, -1, -1}, bbd[6] = {15, -1, 25, -1, -1, -1};
  int cins[6] = {1, 16, 16, 32, 32, 32}, couts[6] = {16, 16, 32, 32, 32, 32};
  const float* cur = p1;
  for (int k2 = 0; k2 < 6; ++k2) {
    int ci = cins[k2], co = couts[k2];
    float* nxt = (k2 & 1) ? rb2 : rb0;
    conv2d_k<<<g((long)B * co * 22 * 200), 256, 0, stream>>>(cur, pf(bw1[k2]), pf(bb1[k2]), nullptr, rb1, B, ci, co, 21, 22, 200, 2, 1, 1);
    const float* addp = cur;
    if (bwd[k2] >= 0) {
      conv2d_k<<<g((long)B * co * 21 * 200), 256, 0, stream>>>(cur, pf(bwd[k2]), pf(bbd[k2]), nullptr, nxt, B, ci, co, 21, 21, 200, 1, 0, 0);
      addp = nxt;
    }
    conv2d_k<<<g((long)B * co * 21 * 200), 256, 0, stream>>>(rb1, pf(bw2[k2]), pf(bb2[k2]), addp, nxt, B, co, co, 22, 21, 200, 2, 0, 0);
    cur = nxt;
  }
  float* xr = (float*)cur;
  ewact_k<<<g((long)B * 32 * 21 * 200), 256, 0, stream>>>(xr, (long)B * 32 * 21 * 200, 1);

  // ---- attention pooling ---------------------------------------------------
  chanlin_k<<<g((long)B * 64 * 4200), 256, 0, stream>>>(xr, pf(38), pf(39), w64, B, 32, 64, 4200, 2);
  chanlin_k<<<g((long)B * 32 * 4200), 256, 0, stream>>>(w64, pf(40), pf(41), w32b, B, 64, 32, 4200, 0);
  es_k<<<g((long)B * 32 * 21), 256, 0, stream>>>(xr, w32b, pf(42), eS);
  et_k<<<g((long)B * 32 * 200), 256, 0, stream>>>(xr, w32b, eT);

  // ---- GATs + top-k pools --------------------------------------------------
  auto run_gat = [&](const float* X, int N, int pb, const float* pw, const float* pbias,
                     int k, float* Out, float* Y) {
    gatatt_k<<<g((long)B * N * N), 256, 0, stream>>>(X, pf(pb + 0), pf(pb + 1), pf(pb + 2), attb, B, N, 32, 32, 0.5f);
    smcol_k<<<g((long)B * N), 256, 0, stream>>>(attb, B, N);
    agg_k<<<g((long)B * N * 32), 256, 0, stream>>>(attb, X, aggb, B, N, 32);
    duallin_k<<<g((long)B * N * 32), 256, 0, stream>>>(aggb, X, pf(pb + 3), pf(pb + 4), pf(pb + 5), pf(pb + 6), Y, B, N, 32, 32, N * 32, 1);
    pool_k<<<B, 128, 0, stream>>>(Y, pw, pbias, Out, N, 32, k);
  };
  run_gat(eS, 21, 43, pf(57), pf(58), 10, outS, gSy);
  run_gat(eT, 200, 50, pf(59), pf(60), 100, outT, gTy);

  // ---- heterogeneous graph blocks -----------------------------------------
  auto run_htrg = [&](const float* x1, int n1, const float* x2, int n2,
                      const float* mast, int mbs, int din, int dout, int pb,
                      float* tOut, float* sOut, float* mOut) {
    int N = n1 + n2; float invt = 0.01f;
    lin_k<<<g((long)B * n1 * din), 256, 0, stream>>>(x1, pf(pb + 0), pf(pb + 1), xc, B, n1, din, din, N, 0, 0);
    lin_k<<<g((long)B * n2 * din), 256, 0, stream>>>(x2, pf(pb + 2), pf(pb + 3), xc, B, n2, din, din, N, n1, 0);
    htrgatt_k<<<g((long)B * N * N), 256, 0, stream>>>(xc, pf(pb + 4), pf(pb + 5), pf(pb + 8), pf(pb + 9), pf(pb + 10), attb, B, N, din, dout, n1, invt);
    smcol_k<<<g((long)B * N), 256, 0, stream>>>(attb, B, N);
    mastatt_k<<<g((long)B * N), 256, 0, stream>>>(xc, mast, mbs, pf(pb + 6), pf(pb + 7), pf(pb + 11), amb, B, N, din, dout, invt);
    mastagg_k<<<B, 64, 0, stream>>>(amb, xc, m1b, N, din);
    duallin_k<<<g((long)B * dout), 256, 0, stream>>>(m1b, mast, pf(pb + 16), pf(pb + 17), pf(pb + 18), pf(pb + 19), mOut, B, 1, din, dout, mbs, 0);
    agg_k<<<g((long)B * N * din), 256, 0, stream>>>(attb, xc, aggb, B, N, din);
    duallin_k<<<g((long)B * N * dout), 256, 0, stream>>>(aggb, xc, pf(pb + 12), pf(pb + 13), pf(pb + 14), pf(pb + 15), ybuf, B, N, din, dout, N * din, 1);
    copyrows_k<<<g((long)B * n1 * dout), 256, 0, stream>>>(ybuf, tOut, B, N, 0, n1, dout);
    copyrows_k<<<g((long)B * n2 * dout), 256, 0, stream>>>(ybuf, sOut, B, N, n1, n2, dout);
  };

  // branch 1
  run_htrg(outT, 100, outS, 10, pf(149), 0, 32, 16, 69, tA, sA, mA);
  pool_k<<<B, 128, 0, stream>>>(sA, pf(61), pf(62), s1p, 10, 16, 5);
  pool_k<<<B, 128, 0, stream>>>(tA, pf(63), pf(64), t1p, 100, 16, 50);
  run_htrg(t1p, 50, s1p, 5, mA, 16, 16, 16, 89, taB, saB, maB);
  add_k<<<g((long)B * 50 * 16), 256, 0, stream>>>(t1p, taB, t1p, (long)B * 50 * 16);
  add_k<<<g((long)B * 5 * 16), 256, 0, stream>>>(s1p, saB, s1p, (long)B * 5 * 16);
  add_k<<<g((long)B * 16), 256, 0, stream>>>(mA, maB, mA, (long)B * 16);
  // branch 2
  run_htrg(outT, 100, outS, 10, pf(150), 0, 32, 16, 109, tA2, sA2, mA2);
  pool_k<<<B, 128, 0, stream>>>(sA2, pf(65), pf(66), s2p, 10, 16, 5);
  pool_k<<<B, 128, 0, stream>>>(tA2, pf(67), pf(68), t2p, 100, 16, 50);
  run_htrg(t2p, 50, s2p, 5, mA2, 16, 16, 16, 129, ta2, sa2, ma2);
  add_k<<<g((long)B * 50 * 16), 256, 0, stream>>>(t2p, ta2, t2p, (long)B * 50 * 16);
  add_k<<<g((long)B * 5 * 16), 256, 0, stream>>>(s2p, sa2, s2p, (long)B * 5 * 16);
  add_k<<<g((long)B * 16), 256, 0, stream>>>(mA2, ma2, mA2, (long)B * 16);

  // merge + head
  max_k<<<g((long)B * 50 * 16), 256, 0, stream>>>(t1p, t2p, t1p, (long)B * 50 * 16);
  max_k<<<g((long)B * 5 * 16), 256, 0, stream>>>(s1p, s2p, s1p, (long)B * 5 * 16);
  max_k<<<g((long)B * 16), 256, 0, stream>>>(mA, mA2, mA, (long)B * 16);
  head_k<<<B, 32, 0, stream>>>(t1p, s1p, mA, pf(151), pf(152), (float*)d_out, 50, 5);
}
